// GATConv_49658411876593
// MI455X (gfx1250) — matrix operand order repaired and checked by reading, not compiled, not run on hardware
//
#include <hip/hip_runtime.h>
#include <hip/hip_bf16.h>
#include <limits.h>

// ---------------------------------------------------------------------------
// GAT forward on gfx1250:
//   h  = feat @ W                (WMMA bf16 -> f32, 16x32 tile per wave)
//   el = h @ attn_l, er = h @ attn_r   (fused epilogue, shfl_xor reduction)
//   e  = leakyrelu(el[src]+er[dst]); edge softmax by dst; out = segsum + bias
// ---------------------------------------------------------------------------

typedef __attribute__((ext_vector_type(16))) __bf16 v16bf;
typedef __attribute__((ext_vector_type(8)))  float  v8f;

#define NEG_SLOPE 0.2f
#define INC 256   // input channels
#define OUTC 32   // output channels

// ---- ordered-int encoding of float for atomicMax-based segment max --------
__device__ __forceinline__ int fenc(float x) {
    int b = __float_as_int(x);
    return b >= 0 ? b : (b ^ 0x7fffffff);
}
__device__ __forceinline__ float fdec(int e) {
    return __int_as_float(e >= 0 ? e : (e ^ 0x7fffffff));
}

// ---- init: out = bias (aggregation accumulates on top), max/denom reset ---
__global__ void gat_init_kernel(float* __restrict__ out,
                                const float* __restrict__ bias,
                                int* __restrict__ mx, float* __restrict__ den,
                                int N) {
    int idx = blockIdx.x * blockDim.x + threadIdx.x;
    if (idx < N * OUTC) out[idx] = bias[idx & (OUTC - 1)];
    if (idx < N) { mx[idx] = INT_MIN; den[idx] = 0.0f; }
}

// ---- GEMM + el/er. Block = 128 threads = 4 waves; wave owns 16 rows. ------
__global__ void __launch_bounds__(128)
gat_gemm_kernel(const float* __restrict__ feat, const float* __restrict__ Wp,
                const float* __restrict__ attn_l, const float* __restrict__ attn_r,
                float* __restrict__ hp, float* __restrict__ elp,
                float* __restrict__ erp, int N) {
    __shared__ __bf16 sWt[OUTC * INC];   // W transposed: sWt[n*256 + k] = W[k][n]

    // stage W (32 KB f32 -> 16 KB bf16), transposed so B fragments are contiguous
    for (int idx = threadIdx.x; idx < INC * OUTC; idx += blockDim.x) {
        int k = idx >> 5, n = idx & 31;
        sWt[n * INC + k] = (__bf16)Wp[idx];
    }
    __syncthreads();

    const int lane = threadIdx.x & 31;
    const int wv   = threadIdx.x >> 5;
    const int half = lane >> 4;        // lane group 0: lanes 0-15, 1: 16-31
    const int n    = lane & 15;
    const int mbase = blockIdx.x * 64 + wv * 16;

    // clamp row for loads so EXEC stays all-ones around WMMA (ISA requirement)
    int m  = mbase + n;
    int mc = m < N ? m : (N - 1);
    const float* rowp = feat + (size_t)mc * INC;

    v8f acc0 = {};   // cols 0..15
    v8f acc1 = {};   // cols 16..31

    for (int kb = 0; kb < INC; kb += 32) {
        if (kb + 32 < INC) __builtin_prefetch(rowp + kb + 32, 0, 0);

        // A fragment: lane half h holds
        // K = kb + h*16 + {0..15} (contiguous, matching B)
        v16bf a;
#pragma unroll
        for (int j = 0; j < 8; ++j) {
            a[j]     = (__bf16)rowp[kb + half * 16 + j];
            a[j + 8] = (__bf16)rowp[kb + half * 16 + 8 + j];
        }
        // B fragments: 16 contiguous bf16 of transposed W per lane
        const v16bf b0 = *(const v16bf*)&sWt[n * INC + kb + half * 16];
        const v16bf b1 = *(const v16bf*)&sWt[(n + 16) * INC + kb + half * 16];

        acc0 = __builtin_amdgcn_wmma_f32_16x16x32_bf16(
            false, a, false, b0, (short)0, acc0, false, false);
        acc1 = __builtin_amdgcn_wmma_f32_16x16x32_bf16(
            false, a, false, b1, (short)0, acc1, false, false);
    }

    // epilogue: store h, fused el/er = h @ attn_{l,r} via 16-lane xor tree
    const float al0 = attn_l[n], al1 = attn_l[n + 16];
    const float ar0 = attn_r[n], ar1 = attn_r[n + 16];
#pragma unroll
    for (int r = 0; r < 8; ++r) {
        // C layout: VGPR r -> M=r (lanes 0-15), M=r+8 (lanes 16-31)
        int row = mbase + half * 8 + r;
        bool ok = row < N;
        float c0 = acc0[r], c1 = acc1[r];
        if (ok) {
            hp[(size_t)row * OUTC + n]      = c0;
            hp[(size_t)row * OUTC + 16 + n] = c1;
        }
        float tl = c0 * al0 + c1 * al1;
        float tr = c0 * ar0 + c1 * ar1;
#pragma unroll
        for (int msk = 8; msk >= 1; msk >>= 1) {
            tl += __shfl_xor(tl, msk, 32);
            tr += __shfl_xor(tr, msk, 32);
        }
        if (ok && n == 0) { elp[row] = tl; erp[row] = tr; }
    }
}

// ---- per-edge score + segment max over dst --------------------------------
__global__ void gat_score_kernel(const float* __restrict__ elp,
                                 const float* __restrict__ erp,
                                 const int* __restrict__ src,
                                 const int* __restrict__ dst,
                                 float* __restrict__ ep, int* __restrict__ mx,
                                 int E) {
    int i = blockIdx.x * blockDim.x + threadIdx.x;
    if (i >= E) return;
    float s = elp[src[i]] + erp[dst[i]];
    s = s > 0.0f ? s : NEG_SLOPE * s;
    ep[i] = s;
    atomicMax(&mx[dst[i]], fenc(s));
}

// ---- ex = exp(e - m[dst]); denom[dst] += ex -------------------------------
__global__ void gat_expsum_kernel(const float* __restrict__ ep,
                                  const int* __restrict__ dst,
                                  const int* __restrict__ mx,
                                  float* __restrict__ den,
                                  float* __restrict__ exv, int E) {
    int i = blockIdx.x * blockDim.x + threadIdx.x;
    if (i >= E) return;
    int d = dst[i];
    float x = __expf(ep[i] - fdec(mx[d]));
    exv[i] = x;
    atomicAdd(&den[d], x);
}

// ---- out[dst] += (ex/denom[dst]) * h[src]; 8 threads/edge, float4 each ----
__global__ void gat_aggregate_kernel(const float* __restrict__ exv,
                                     const float* __restrict__ den,
                                     const int* __restrict__ src,
                                     const int* __restrict__ dst,
                                     const float* __restrict__ hp,
                                     float* __restrict__ out, int E) {
    long long gid = (long long)blockIdx.x * blockDim.x + threadIdx.x;
    int i = (int)(gid >> 3);
    if (i >= E) return;
    int c = ((int)gid & 7) * 4;
    int d = dst[i];
    float alpha = exv[i] / den[d];
    float4 hv = *(const float4*)&hp[(size_t)src[i] * OUTC + c];
    float* o = out + (size_t)d * OUTC + c;
    atomicAdd(o + 0, alpha * hv.x);
    atomicAdd(o + 1, alpha * hv.y);
    atomicAdd(o + 2, alpha * hv.z);
    atomicAdd(o + 3, alpha * hv.w);
}

// ---------------------------------------------------------------------------
extern "C" void kernel_launch(void* const* d_in, const int* in_sizes, int n_in,
                              void* d_out, int out_size, void* d_ws, size_t ws_size,
                              hipStream_t stream) {
    const float* feat   = (const float*)d_in[0];
    const float* W      = (const float*)d_in[1];
    const float* attn_l = (const float*)d_in[2];
    const float* attn_r = (const float*)d_in[3];
    const float* bias   = (const float*)d_in[4];
    const int*   src    = (const int*)d_in[5];
    const int*   dst    = (const int*)d_in[6];

    const int N = in_sizes[0] / INC;   // 100000
    const int E = in_sizes[5];         // 1600000
    float* out = (float*)d_out;

    // workspace carve-out (~27 MB): h, el, er, mx, denom, e, ex
    float* h   = (float*)d_ws;
    float* el  = h + (size_t)N * OUTC;
    float* er  = el + N;
    int*   mx  = (int*)(er + N);
    float* den = (float*)(mx + N);
    float* e   = den + N;
    float* ex  = e + E;

    gat_init_kernel<<<(N * OUTC + 255) / 256, 256, 0, stream>>>(out, bias, mx, den, N);
    gat_gemm_kernel<<<(N + 63) / 64, 128, 0, stream>>>(feat, W, attn_l, attn_r,
                                                       h, el, er, N);
    gat_score_kernel<<<(E + 255) / 256, 256, 0, stream>>>(el, er, src, dst, e, mx, E);
    gat_expsum_kernel<<<(E + 255) / 256, 256, 0, stream>>>(e, dst, mx, den, ex, E);
    long long aggThreads = (long long)E * 8;
    gat_aggregate_kernel<<<(unsigned)((aggThreads + 255) / 256), 256, 0, stream>>>(
        ex, den, src, dst, h, out, E);
}
